// PointTransformerNorm_35046933135992
// MI455X (gfx1250) — compile-verified, matrix operand order checked by reading, not run on hardware
//
#include <hip/hip_runtime.h>

typedef _Float16 half16 __attribute__((ext_vector_type(16)));
typedef float    float8 __attribute__((ext_vector_type(8)));
typedef int      i4v    __attribute__((ext_vector_type(4)));

#define NPTS 2048
#define BATCH 8
#define EPS_BN 1e-5f
#define LDSS 40  // padded LDS row stride in halfs (80B -> conflict-free, 16B aligned)

#if defined(__HIP_DEVICE_COMPILE__) && __has_builtin(__builtin_amdgcn_global_load_async_to_lds_b128) && __has_builtin(__builtin_amdgcn_s_wait_asynccnt)
#define USE_ASYNC_LDS 1
#else
#define USE_ASYNC_LDS 0
#endif

// ---------------------------------------------------------------------------
// Epilogue descriptor for the generic WMMA GEMM. All f32 activation planes
// ([C, NPTS]) share ld32 = NPTS.
// ---------------------------------------------------------------------------
struct EpiP {
  const float* bias;      // per-output-channel bias (nullptr = none)
  const float* g;         // BN gamma (nullptr = no BN)
  const float* b;         // BN beta
  const float* m;         // BN mean
  const float* v;         // BN var
  const float* sub_base;  // if set: val = sub_base[c][n] - val   (x - x_r)
  const float* res_base;  // if set: val = res_base[c][n] + val   (residual)
  float*    out_f32;      // optional f32 [C, NPTS] store
  int       ld32;
  _Float16* out_h;        // optional f16 [C, ldh] store (direct layout)
  int       ldh;
  _Float16* out_hT;       // optional f16 [N, ldhT] store (transposed layout)
  int       ldhT;
  int       hT_off;       // column offset inside transposed buffer
  int       act;          // 0 none, 1 relu, 2 leaky-relu(0.2)
};

// 16x32 f16 fragment from global (A operand; row-major, K contiguous).
// ISA 16-bit A 16x32: lane L -> row L%16, half = L/16;
// VGPR j<4 holds K = 2j + 8*half (+1); VGPR j>=4 holds K = 16 + 2(j-4) + 8*half.
__device__ __forceinline__ half16 load_frag_g(const _Float16* __restrict__ base,
                                              int ld, int row, int k0, int half_id) {
  union { half16 v; unsigned u[8]; } f;
  const unsigned* p = (const unsigned*)(base + (size_t)row * ld + k0);
#pragma unroll
  for (int j = 0; j < 8; ++j) {
    int kk = ((j < 4) ? (2 * j) : (16 + 2 * (j - 4))) + 8 * half_id;
    f.u[j] = p[kk >> 1];
  }
  return f.v;
}

// Same fragment pattern from an LDS row (32 halfs at `row`).
__device__ __forceinline__ half16 load_frag_lds(const _Float16* row, int half_id) {
  union { half16 v; unsigned u[8]; } f;
  const unsigned* p = (const unsigned*)row;
#pragma unroll
  for (int j = 0; j < 8; ++j) {
    int kk = ((j < 4) ? (2 * j) : (16 + 2 * (j - 4))) + 8 * half_id;
    f.u[j] = p[kk >> 1];
  }
  return f.v;
}

// Stage 16 bytes global -> LDS (async DMA path on CDNA5 when available).
__device__ __forceinline__ void stage16(const _Float16* src, _Float16* dst) {
#if USE_ASYNC_LDS
  _Float16* s_nc = const_cast<_Float16*>(src);
  __builtin_amdgcn_global_load_async_to_lds_b128(
      (__attribute__((address_space(1))) i4v*)s_nc,
      (__attribute__((address_space(3))) i4v*)dst, 0, 0);
#else
  *(i4v*)dst = *(const i4v*)src;
#endif
}

__device__ __forceinline__ void stage_wait() {
#if USE_ASYNC_LDS
  __builtin_amdgcn_s_wait_asynccnt(0);
#endif
}

// ---------------------------------------------------------------------------
// Generic GEMM: Out[M, NPTS] = A[M,K] (f16 row-major, lda) @ B[K,NPTS],
// B supplied transposed: Bt[NPTS, K] (f16 row-major, ldbt).
// Block = 256 threads (8 waves). Block computes a 128(M) x 64(N) tile:
// the 64x32 B k-slab is staged once into double-buffered LDS and shared by
// all 8 waves; each wave runs 4 WMMAs/k-step on its own 16-row A fragment.
// grid = (ceil(M/128), NPTS/64).
// ---------------------------------------------------------------------------
__global__ __launch_bounds__(256) void gemm_wmma_f16(
    const _Float16* __restrict__ A, int lda,
    const _Float16* __restrict__ Bt, int ldbt,
    int M, int K, EpiP ep) {
  __shared__ _Float16 Bs[2][64 * LDSS];

  const int tid     = threadIdx.x;
  const int wave    = tid >> 5;
  const int lane    = tid & 31;
  const int half_id = lane >> 4;
  const int l16     = lane & 15;
  const int m0      = (blockIdx.x * 8 + wave) * 16;
  const int n0      = blockIdx.y * 64;
  const bool active = (m0 < M);

  // staging assignment: each thread moves 16B (8 halfs) per k-step
  const int scol = tid >> 2;        // 0..63 : B^T row (column of B)
  const int sseg = (tid & 3) * 8;   // 0,8,16,24 halfs within the 32-k slab
  const _Float16* sgp = Bt + (size_t)(n0 + scol) * ldbt + sseg;
  _Float16* sd0 = &Bs[0][scol * LDSS + sseg];
  _Float16* sd1 = &Bs[1][scol * LDSS + sseg];

  float8 acc0 = {}, acc1 = {}, acc2 = {}, acc3 = {};

  const int nk = K >> 5;
  stage16(sgp, sd0);

  for (int i = 0; i < nk; ++i) {
    stage_wait();
    __syncthreads();  // staged slab i visible; previous compute done
    if (i + 1 < nk) stage16(sgp + (size_t)(i + 1) * 32, ((i + 1) & 1) ? sd1 : sd0);
    if (active) {
      const _Float16* bb = &Bs[i & 1][0];
      half16 a  = load_frag_g(A, lda, m0 + l16, i * 32, half_id);
      half16 b0 = load_frag_lds(bb + (l16 +  0) * LDSS, half_id);
      half16 b1 = load_frag_lds(bb + (l16 + 16) * LDSS, half_id);
      half16 b2 = load_frag_lds(bb + (l16 + 32) * LDSS, half_id);
      half16 b3 = load_frag_lds(bb + (l16 + 48) * LDSS, half_id);
      acc0 = __builtin_amdgcn_wmma_f32_16x16x32_f16(false, a, false, b0, (short)0, acc0, false, false);
      acc1 = __builtin_amdgcn_wmma_f32_16x16x32_f16(false, a, false, b1, (short)0, acc1, false, false);
      acc2 = __builtin_amdgcn_wmma_f32_16x16x32_f16(false, a, false, b2, (short)0, acc2, false, false);
      acc3 = __builtin_amdgcn_wmma_f32_16x16x32_f16(false, a, false, b3, (short)0, acc3, false, false);
    }
  }

  if (!active) return;

  union { float8 v; float f[8]; } rr[4];
  rr[0].v = acc0; rr[1].v = acc1; rr[2].v = acc2; rr[3].v = acc3;

#pragma unroll
  for (int t = 0; t < 4; ++t) {
    const int col = n0 + 16 * t + l16;
#pragma unroll
    for (int r = 0; r < 8; ++r) {
      const int crow = m0 + r + 8 * half_id;  // C/D layout: VGPR r -> M = r + 8*(lane/16)
      float val = rr[t].f[r];
      if (ep.bias)     val += ep.bias[crow];
      if (ep.sub_base) val  = ep.sub_base[(size_t)crow * ep.ld32 + col] - val;
      if (ep.g) {
        float s = ep.g[crow] * rsqrtf(ep.v[crow] + EPS_BN);
        val = s * (val - ep.m[crow]) + ep.b[crow];
      }
      if      (ep.act == 1) val = fmaxf(val, 0.0f);
      else if (ep.act == 2) val = (val > 0.0f) ? val : 0.2f * val;
      if (ep.res_base) val += ep.res_base[(size_t)crow * ep.ld32 + col];
      if (ep.out_f32) ep.out_f32[(size_t)crow * ep.ld32 + col] = val;
      if (ep.out_h)   ep.out_h[(size_t)crow * ep.ldh + col] = (_Float16)val;
      if (ep.out_hT)  ep.out_hT[(size_t)col * ep.ldhT + ep.hT_off + crow] = (_Float16)val;
    }
  }
}

// ---------------------------------------------------------------------------
// f32 -> f16 weight conversion
// ---------------------------------------------------------------------------
__global__ void cvt_f32_to_f16(const float* __restrict__ in, _Float16* __restrict__ out, int n) {
  int i = blockIdx.x * blockDim.x + threadIdx.x;
  if (i < n) out[i] = (_Float16)in[i];
}

// conv1 (K=3) + BN + ReLU, writes transposed f16 [N,256]
__global__ void conv1_bn_relu(const float* __restrict__ x, const float* __restrict__ w1,
                              const float* __restrict__ g, const float* __restrict__ b,
                              const float* __restrict__ m, const float* __restrict__ v,
                              _Float16* __restrict__ X1T, int bidx) {
  int idx = blockIdx.x * blockDim.x + threadIdx.x;  // over NPTS*256
  int n = idx >> 8, c = idx & 255;
  const float* p = x + ((size_t)bidx * NPTS + n) * 3;
  float acc = w1[c * 3 + 0] * p[0] + w1[c * 3 + 1] * p[1] + w1[c * 3 + 2] * p[2];
  float s = g[c] * rsqrtf(v[c] + EPS_BN);
  acc = fmaxf(s * (acc - m[c]) + b[c], 0.0f);
  X1T[(size_t)n * 256 + c] = (_Float16)acc;
}

// row-wise softmax of S[NPTS][NPTS], in place
__global__ __launch_bounds__(256) void softmax_rows(float* __restrict__ S) {
  __shared__ float red[256];
  const int tid = threadIdx.x;
  float* row = S + (size_t)blockIdx.x * NPTS;
  float vals[NPTS / 256];
  float vmax = -3.0e38f;
#pragma unroll
  for (int i = 0; i < NPTS / 256; ++i) { vals[i] = row[tid + i * 256]; vmax = fmaxf(vmax, vals[i]); }
  red[tid] = vmax; __syncthreads();
  for (int s = 128; s > 0; s >>= 1) { if (tid < s) red[tid] = fmaxf(red[tid], red[tid + s]); __syncthreads(); }
  vmax = red[0]; __syncthreads();
  float vsum = 0.0f;
#pragma unroll
  for (int i = 0; i < NPTS / 256; ++i) { vals[i] = __expf(vals[i] - vmax); vsum += vals[i]; }
  red[tid] = vsum; __syncthreads();
  for (int s = 128; s > 0; s >>= 1) { if (tid < s) red[tid] += red[tid + s]; __syncthreads(); }
  float inv = 1.0f / red[0];
#pragma unroll
  for (int i = 0; i < NPTS / 256; ++i) row[tid + i * 256] = vals[i] * inv;
}

// column sums of attn
__global__ void attn_colsum(const float* __restrict__ A, float* __restrict__ cs) {
  int m = blockIdx.x * blockDim.x + threadIdx.x;
  float s = 0.0f;
  for (int n = 0; n < NPTS; ++n) s += A[(size_t)n * NPTS + m];
  cs[m] = s;
}

// attnT[m][n] = f16( attn[n][m] / (1e-9 + colsum[m]) ), LDS-tiled transpose
__global__ void attn_transpose_scale(const float* __restrict__ A, const float* __restrict__ cs,
                                     _Float16* __restrict__ AT) {
  __shared__ float tile[16][17];
  const int m0 = blockIdx.x * 16, n0 = blockIdx.y * 16;
  tile[threadIdx.y][threadIdx.x] = A[(size_t)(n0 + threadIdx.y) * NPTS + m0 + threadIdx.x];
  __syncthreads();
  int m = m0 + threadIdx.y;
  float sc = 1.0f / (1e-9f + cs[m]);
  AT[(size_t)m * NPTS + n0 + threadIdx.x] = (_Float16)(tile[threadIdx.x][threadIdx.y] * sc);
}

// gf[c] = max_n Xf[c][n] (c<1024); gf[1024+c] = mean_n Xf[c][n]
__global__ void pool_max_avg(const float* __restrict__ Xf, float* __restrict__ gf) {
  int c = blockIdx.x * blockDim.x + threadIdx.x;  // 0..1023
  const float* row = Xf + (size_t)c * NPTS;
  float vmax = -3.0e38f, vsum = 0.0f;
  for (int n = 0; n < NPTS; ++n) { float x = row[n]; vmax = fmaxf(vmax, x); vsum += x; }
  gf[c] = vmax;
  gf[1024 + c] = vsum * (1.0f / NPTS);
}

// Gt[n][c] = f16(gf[n]) for c < 1024 (the reference's broadcast quirk)
__global__ void gt_fill(const float* __restrict__ gf, _Float16* __restrict__ Gt) {
  int idx = blockIdx.x * blockDim.x + threadIdx.x;  // NPTS*1024
  int n = idx >> 10, c = idx & 1023;
  Gt[(size_t)n * 2048 + c] = (_Float16)gf[n];
}

// final 3-channel conv + L2 normalize; out[b][n][c3]
__global__ void head_normalize(const _Float16* __restrict__ X2T, const float* __restrict__ w3,
                               const float* __restrict__ b3, float* __restrict__ out, int bidx) {
  int n = blockIdx.x * blockDim.x + threadIdx.x;  // 0..NPTS-1
  const _Float16* xr = X2T + (size_t)n * 256;
  float o0 = b3[0], o1 = b3[1], o2 = b3[2];
  for (int c = 0; c < 256; ++c) {
    float xv = (float)xr[c];
    o0 += w3[c] * xv; o1 += w3[256 + c] * xv; o2 += w3[512 + c] * xv;
  }
  float nn = fmaxf(sqrtf(o0 * o0 + o1 * o1 + o2 * o2), 1e-12f);
  float inv = 1.0f / nn;
  float* po = out + ((size_t)bidx * NPTS + n) * 3;
  po[0] = o0 * inv; po[1] = o1 * inv; po[2] = o2 * inv;
}

// ---------------------------------------------------------------------------
// Host side
// ---------------------------------------------------------------------------
struct BNPtrs { const float *g, *b, *m, *v; };

static inline void launch_gemm(const _Float16* A, int lda, const _Float16* Bt, int ldbt,
                               int M, int K, const EpiP& ep, hipStream_t s) {
  dim3 grid((M + 127) / 128, NPTS / 64);
  gemm_wmma_f16<<<grid, 256, 0, s>>>(A, lda, Bt, ldbt, M, K, ep);
}

static inline void cvt(const float* in, _Float16* out, int n, hipStream_t s) {
  cvt_f32_to_f16<<<(n + 255) / 256, 256, 0, s>>>(in, out, n);
}

extern "C" void kernel_launch(void* const* d_in, const int* in_sizes, int n_in,
                              void* d_out, int out_size, void* d_ws, size_t ws_size,
                              hipStream_t stream) {
  (void)in_sizes; (void)n_in; (void)out_size; (void)ws_size;

  // ---- unpack inputs (setup_inputs dict order, leaves flattened) ----
  int i = 0;
  auto F = [&](void) { return (const float*)d_in[i++]; };
  const float* x_in = F();
  const float* w1 = F();
  BNPtrs bn1{F(), F(), F(), F()};
  const float* w2 = F();
  BNPtrs bn2{F(), F(), F(), F()};
  const float *wqk[4], *wv[4], *bv[4], *wt[4], *bt[4];
  BNPtrs sabn[4];
  for (int l = 0; l < 4; ++l) {
    wqk[l] = F(); wv[l] = F(); bv[l] = F(); wt[l] = F(); bt[l] = F();
    sabn[l] = BNPtrs{F(), F(), F(), F()};
  }
  const float* wf = F();
  BNPtrs bnf{F(), F(), F(), F()};
  const float* ws1 = F();
  const float* bs1 = F();
  BNPtrs bns1{F(), F(), F(), F()};
  const float* ws2 = F();
  const float* bs2 = F();
  BNPtrs bns2{F(), F(), F(), F()};
  const float* ws3 = F();
  const float* bs3 = F();
  float* out = (float*)d_out;

  // ---- workspace carve-up ----
  char* p = (char*)d_ws;
  auto alloc = [&](size_t bytes) -> void* {
    void* r = (void*)p;
    p += (bytes + 255) & ~(size_t)255;
    return r;
  };
  _Float16* W2h  = (_Float16*)alloc(256 * 256 * 2);
  _Float16 *Wqkh[4], *Wvh[4], *Wth[4];
  for (int l = 0; l < 4; ++l) {
    Wqkh[l] = (_Float16*)alloc(64 * 256 * 2);
    Wvh[l]  = (_Float16*)alloc(256 * 256 * 2);
    Wth[l]  = (_Float16*)alloc(256 * 256 * 2);
  }
  _Float16* Wfh  = (_Float16*)alloc(1024 * 1024 * 2);
  _Float16* Ws1h = (_Float16*)alloc(512 * 2048 * 2);
  _Float16* Ws2h = (_Float16*)alloc(256 * 512 * 2);

  _Float16* X1Th   = (_Float16*)alloc((size_t)NPTS * 256 * 2);
  _Float16* XTh    = (_Float16*)alloc((size_t)NPTS * 256 * 2);
  float*    Xf32   = (float*)   alloc((size_t)256 * NPTS * 4);
  _Float16* FTh    = (_Float16*)alloc((size_t)NPTS * 1024 * 2);
  _Float16* YTh    = (_Float16*)alloc((size_t)NPTS * 64 * 2);
  _Float16* Vh     = (_Float16*)alloc((size_t)256 * NPTS * 2);
  float*    Smat   = (float*)   alloc((size_t)NPTS * NPTS * 4);
  float*    csum   = (float*)   alloc((size_t)NPTS * 4);
  _Float16* attnTh = (_Float16*)alloc((size_t)NPTS * NPTS * 2);
  _Float16* diffTh = (_Float16*)alloc((size_t)NPTS * 256 * 2);
  float*    Xff32  = (float*)   alloc((size_t)1024 * NPTS * 4);
  _Float16* Gth    = (_Float16*)alloc((size_t)NPTS * 2048 * 2);
  float*    gf     = (float*)   alloc((size_t)2048 * 4);
  _Float16* Xs1Th  = (_Float16*)alloc((size_t)NPTS * 512 * 2);
  _Float16* Xs2Th  = (_Float16*)alloc((size_t)NPTS * 256 * 2);

  // ---- convert weights to f16 once ----
  cvt(w2, W2h, 256 * 256, stream);
  for (int l = 0; l < 4; ++l) {
    cvt(wqk[l], Wqkh[l], 64 * 256, stream);
    cvt(wv[l],  Wvh[l],  256 * 256, stream);
    cvt(wt[l],  Wth[l],  256 * 256, stream);
  }
  cvt(wf,  Wfh,  1024 * 1024, stream);
  cvt(ws1, Ws1h, 512 * 2048, stream);
  cvt(ws2, Ws2h, 256 * 512, stream);

  // ---- per-batch pipeline ----
  for (int b = 0; b < BATCH; ++b) {
    // conv1 + bn + relu -> X1T
    conv1_bn_relu<<<(NPTS * 256) / 256, 256, 0, stream>>>(x_in, w1, bn1.g, bn1.b, bn1.m, bn1.v, X1Th, b);

    // conv2 + bn + relu -> Xf32, XTh
    {
      EpiP ep{}; ep.g = bn2.g; ep.b = bn2.b; ep.m = bn2.m; ep.v = bn2.v;
      ep.act = 1; ep.out_f32 = Xf32; ep.ld32 = NPTS; ep.out_hT = XTh; ep.ldhT = 256;
      launch_gemm(W2h, 256, X1Th, 256, 256, 256, ep, stream);
    }

    for (int l = 0; l < 4; ++l) {
      const _Float16* inT = (l == 0) ? XTh : (FTh + (l - 1) * 256);
      const int in_ld = (l == 0) ? 256 : 1024;

      // Y = wqk @ x -> YT [N,64]
      { EpiP ep{}; ep.ld32 = NPTS; ep.out_hT = YTh; ep.ldhT = 64;
        launch_gemm(Wqkh[l], 256, inT, in_ld, 64, 256, ep, stream); }
      // V = wv @ x + bv -> Vh [256,N]
      { EpiP ep{}; ep.bias = bv[l]; ep.ld32 = NPTS; ep.out_h = Vh; ep.ldh = NPTS;
        launch_gemm(Wvh[l], 256, inT, in_ld, 256, 256, ep, stream); }
      // S = Y^T Y  (A = YT, Bt = YT)
      { EpiP ep{}; ep.out_f32 = Smat; ep.ld32 = NPTS;
        launch_gemm(YTh, 64, YTh, 64, NPTS, 64, ep, stream); }
      // softmax rows, column renorm, transposed f16 attn
      softmax_rows<<<NPTS, 256, 0, stream>>>(Smat);
      attn_colsum<<<NPTS / 256, 256, 0, stream>>>(Smat, csum);
      attn_transpose_scale<<<dim3(NPTS / 16, NPTS / 16), dim3(16, 16), 0, stream>>>(Smat, csum, attnTh);
      // x_r = V @ attn ; diffT = f16(x - x_r)
      { EpiP ep{}; ep.sub_base = Xf32; ep.ld32 = NPTS; ep.out_hT = diffTh; ep.ldhT = 256;
        launch_gemm(Vh, NPTS, attnTh, NPTS, 256, NPTS, ep, stream); }
      // x = x + relu(bn(wt @ diff + bt)); write f32 x and xT slice of concat
      { EpiP ep{}; ep.bias = bt[l];
        ep.g = sabn[l].g; ep.b = sabn[l].b; ep.m = sabn[l].m; ep.v = sabn[l].v;
        ep.act = 1; ep.res_base = Xf32; ep.out_f32 = Xf32; ep.ld32 = NPTS;
        ep.out_hT = FTh; ep.ldhT = 1024; ep.hT_off = l * 256;
        launch_gemm(Wth[l], 256, diffTh, 256, 256, 256, ep, stream); }
    }

    // xf = leaky_relu(bnf(wf @ concat)) -> Xff32 and Gt[:,1024:]
    { EpiP ep{}; ep.g = bnf.g; ep.b = bnf.b; ep.m = bnf.m; ep.v = bnf.v;
      ep.act = 2; ep.out_f32 = Xff32; ep.ld32 = NPTS;
      ep.out_hT = Gth; ep.ldhT = 2048; ep.hT_off = 1024;
      launch_gemm(Wfh, 1024, FTh, 1024, 1024, 1024, ep, stream); }

    pool_max_avg<<<4, 256, 0, stream>>>(Xff32, gf);
    gt_fill<<<(NPTS * 1024) / 256, 256, 0, stream>>>(gf, Gth);

    // ws1 + bias + bn + relu -> Xs1T
    { EpiP ep{}; ep.bias = bs1; ep.g = bns1.g; ep.b = bns1.b; ep.m = bns1.m; ep.v = bns1.v;
      ep.act = 1; ep.ld32 = NPTS; ep.out_hT = Xs1Th; ep.ldhT = 512;
      launch_gemm(Ws1h, 2048, Gth, 2048, 512, 2048, ep, stream); }
    // ws2 + bias + bn + relu -> Xs2T
    { EpiP ep{}; ep.bias = bs2; ep.g = bns2.g; ep.b = bns2.b; ep.m = bns2.m; ep.v = bns2.v;
      ep.act = 1; ep.ld32 = NPTS; ep.out_hT = Xs2Th; ep.ldhT = 256;
      launch_gemm(Ws2h, 512, Xs1Th, 512, 256, 512, ep, stream); }

    // final head + F.normalize
    head_normalize<<<NPTS / 256, 256, 0, stream>>>(Xs2Th, ws3, bs3, out, b);
  }
}
